// Cross_Modal_Attention_19026705121913
// MI455X (gfx1250) — matrix operand order repaired and checked by reading, not compiled, not run on hardware
//
#include <hip/hip_runtime.h>
#include <hip/hip_bf16.h>

// ---- problem constants ----
#define BATCH 2
#define SEQ   2048
#define EMB   512
#define HEADS 8
#define HDIM  64
#define F3    (3*EMB)          // 1536
#define MTOT  (BATCH*SEQ)      // 4096

typedef __attribute__((ext_vector_type(16))) __bf16 v16bf;
typedef __attribute__((ext_vector_type(8)))  float  v8f;

static __device__ inline v8f wmma_bf16(v16bf a, v16bf b, v8f c) {
  // D = A(16x32 bf16) x B(32x16 bf16) + C(16x16 f32)
  return __builtin_amdgcn_wmma_f32_16x16x32_bf16(false, a, false, b, (short)0, c,
                                                 false, false);
}

// A fragment (16x32 bf16, wave32): lane covers row m = lane&15, half = lane>>4.
// a[0..7]  = A[m][k0 +      half*8 + j]
// a[8..15] = A[m][k0 + 16 + half*8 + j]
static __device__ inline v16bf load_afrag(const __bf16* rowptr, int k0, int hi) {
  v16bf a;
  const __bf16* p0 = rowptr + k0 + hi * 8;
  const __bf16* p1 = rowptr + k0 + 16 + hi * 8;
#pragma unroll
  for (int j = 0; j < 8; ++j) { a[j] = p0[j]; a[8 + j] = p1[j]; }
  return a;
}

// B fragment (32x16 bf16): lane covers col n = lane&15; b[0..7] = B[k0+half*8+j][n],
// b[8..15] = B[k0+16+half*8+j][n] (W stored row-major as W[n][k]).
static __device__ inline v16bf load_bfrag(const __bf16* rowptr, int k0, int hi) {
  v16bf b;
  const __bf16* p0 = rowptr + k0 + hi * 8;
  const __bf16* p1 = rowptr + k0 + 16 + hi * 8;
#pragma unroll
  for (int j = 0; j < 8; ++j) { b[j] = p0[j]; b[8 + j] = p1[j]; }
  return b;
}

// ---------------- kernel 0: fp32 -> bf16 packing ----------------
__global__ void pack_bf16(const float* __restrict__ x1, const float* __restrict__ x2,
                          const float* __restrict__ qkvw, const float* __restrict__ outw,
                          __bf16* __restrict__ xbf, __bf16* __restrict__ wbf,
                          __bf16* __restrict__ owbf) {
  const size_t NX = (size_t)MTOT * EMB;        // 2,097,152
  const size_t NW = (size_t)F3 * EMB;          // 786,432
  const size_t NO = (size_t)EMB * EMB;         // 262,144
  size_t i = (size_t)blockIdx.x * blockDim.x + threadIdx.x;
  if (i < NX)                xbf[i] = (__bf16)x1[i];
  else if (i < 2 * NX)       xbf[i] = (__bf16)x2[i - NX];
  else if (i < 2 * NX + NW)  wbf[i - 2 * NX] = (__bf16)qkvw[i - 2 * NX];
  else if (i < 2 * NX + NW + NO) owbf[i - 2 * NX - NW] = (__bf16)outw[i - 2 * NX - NW];
}

// ---------------- kernel 1: fused QKV projection ----------------
// 32(M) x 64(N) register tile per wave: 2 A-frags, 4 B-frags, 8 f32 accumulators.
// The 64-aligned N window lies inside one (head, which) segment -> uniform scatter.
// xbf:[2][4096][512]  wbf:[1536][512] -> Q,K:[2][2][8][2048][64], Vt:[2][2][8][64][2048]
__global__ __launch_bounds__(32) void qkv_gemm(const __bf16* __restrict__ xbf,
                                               const __bf16* __restrict__ wbf,
                                               const float* __restrict__ qkv_b,
                                               __bf16* __restrict__ Qb,
                                               __bf16* __restrict__ Kb,
                                               __bf16* __restrict__ Vt) {
  const int lane = threadIdx.x;
  const int col  = lane & 15;
  const int hi   = lane >> 4;

  int t = blockIdx.x;                      // inp*128*24 + mt*24 + nt
  const int nt  = t % (F3 / 64); t /= (F3 / 64);
  const int mt  = t % (MTOT / 32);
  const int inp = t / (MTOT / 32);
  const int m0 = mt * 32, n0 = nt * 64;

  const __bf16* A0 = xbf + (size_t)inp * MTOT * EMB + (size_t)(m0 + col) * EMB;
  const __bf16* A1 = A0 + (size_t)16 * EMB;
  const __bf16* Bw = wbf + (size_t)(n0 + col) * EMB;

  v8f acc[2][4] = {};
  for (int k0 = 0; k0 < EMB; k0 += 32) {
    v16bf a0 = load_afrag(A0, k0, hi);
    v16bf a1 = load_afrag(A1, k0, hi);
#pragma unroll
    for (int sub = 0; sub < 4; ++sub) {
      v16bf b = load_bfrag(Bw + (size_t)sub * 16 * EMB, k0, hi);
      acc[0][sub] = wmma_bf16(a0, b, acc[0][sub]);
      acc[1][sub] = wmma_bf16(a1, b, acc[1][sub]);
    }
  }

  const int seg   = n0 >> 6;               // n0 is 64-aligned
  const int h     = seg / 3;
  const int which = seg % 3;               // 0=Q 1=K 2=V (uniform per block)
  float bias[4];
#pragma unroll
  for (int sub = 0; sub < 4; ++sub) bias[sub] = qkv_b[n0 + sub * 16 + col];

#pragma unroll
  for (int mtile = 0; mtile < 2; ++mtile) {
#pragma unroll
    for (int r = 0; r < 8; ++r) {
      const int m  = m0 + mtile * 16 + r + hi * 8;   // row in [0,4096)
      const int bb = m >> 11;
      const int l  = m & (SEQ - 1);
      const size_t bh = (((size_t)inp * BATCH + bb) * HEADS + h);
#pragma unroll
      for (int sub = 0; sub < 4; ++sub) {
        const int d   = sub * 16 + col;
        const float v = acc[mtile][sub][r] + bias[sub];
        if (which == 0)      Qb[(bh * SEQ + l) * HDIM + d] = (__bf16)(v * 0.125f);
        else if (which == 1) Kb[(bh * SEQ + l) * HDIM + d] = (__bf16)v;
        else                 Vt[(bh * HDIM + d) * SEQ + l] = (__bf16)v;
      }
    }
  }
}

// ---------------- kernel 2: flash cross-attention ----------------
// pair 0: Q from input0, K/V from input1 ; pair 1: Q from input1, K/V from input0
// Ob: [2][4096][512] bf16 (pair, b*L+l, h*64+d)
__global__ __launch_bounds__(32) void cross_attn(const __bf16* __restrict__ Qb,
                                                 const __bf16* __restrict__ Kb,
                                                 const __bf16* __restrict__ Vt,
                                                 __bf16* __restrict__ Ob) {
  __shared__ __bf16 Plds[16 * 32];
  const int lane = threadIdx.x;
  const int col  = lane & 15;
  const int hi   = lane >> 4;

  int t = blockIdx.x;                      // pair*2*8*128 + b*8*128 + h*128 + qt
  const int qt   = t % (SEQ / 16); t /= (SEQ / 16);
  const int h    = t % HEADS;      t /= HEADS;
  const int bb   = t % BATCH;
  const int pair = t / BATCH;
  const int qi = pair, ki = 1 - pair;

  const __bf16* Qp = Qb + (((size_t)qi * BATCH + bb) * HEADS + h) * SEQ * HDIM;
  const __bf16* Kp = Kb + (((size_t)ki * BATCH + bb) * HEADS + h) * SEQ * HDIM;
  const __bf16* Vp = Vt + (((size_t)ki * BATCH + bb) * HEADS + h) * HDIM * SEQ;

  // Q fragments for d = 0..63 (rows qt*16 .. +15); Q pre-scaled by 1/sqrt(D)
  const __bf16* qrow = Qp + (size_t)(qt * 16 + col) * HDIM;
  v16bf qa0 = load_afrag(qrow, 0, hi);
  v16bf qa1 = load_afrag(qrow, 32, hi);

  float mrun[8], lrun[8];
  v8f o[4] = {};
#pragma unroll
  for (int r = 0; r < 8; ++r) { mrun[r] = -1.0e30f; lrun[r] = 0.0f; }

  for (int kb = 0; kb < SEQ; kb += 32) {
    // prefetch next key/value block (global_prefetch_b8)
    if (kb + 32 < SEQ) {
      __builtin_prefetch(Kp + (size_t)(kb + 32 + lane) * HDIM, 0, 0);
      __builtin_prefetch(Vp + (size_t)(col)*SEQ + kb + 32, 0, 0);
    }
    // S0 covers keys kb..kb+15, S1 covers kb+16..kb+31 ; gemm-K dim = d (64)
    const __bf16* krow0 = Kp + (size_t)(kb + col) * HDIM;
    const __bf16* krow1 = Kp + (size_t)(kb + 16 + col) * HDIM;
    v8f s0 = {}, s1 = {};
    s0 = wmma_bf16(qa0, load_bfrag(krow0, 0, hi), s0);
    s0 = wmma_bf16(qa1, load_bfrag(krow0, 32, hi), s0);
    s1 = wmma_bf16(qa0, load_bfrag(krow1, 0, hi), s1);
    s1 = wmma_bf16(qa1, load_bfrag(krow1, 32, hi), s1);

    // online softmax: row r+8*hi lives across the 16 lanes of each half-wave
#pragma unroll
    for (int r = 0; r < 8; ++r) {
      float mx = fmaxf(s0[r], s1[r]);
      mx = fmaxf(mx, __shfl_xor(mx, 1, 32));
      mx = fmaxf(mx, __shfl_xor(mx, 2, 32));
      mx = fmaxf(mx, __shfl_xor(mx, 4, 32));
      mx = fmaxf(mx, __shfl_xor(mx, 8, 32));
      const float nm = fmaxf(mrun[r], mx);
      const float al = __expf(mrun[r] - nm);
      const float p0 = __expf(s0[r] - nm);
      const float p1 = __expf(s1[r] - nm);
      s0[r] = p0; s1[r] = p1;
      float sum = p0 + p1;
      sum += __shfl_xor(sum, 1, 32);
      sum += __shfl_xor(sum, 2, 32);
      sum += __shfl_xor(sum, 4, 32);
      sum += __shfl_xor(sum, 8, 32);
      lrun[r] = lrun[r] * al + sum;
      mrun[r] = nm;
      o[0][r] *= al; o[1][r] *= al; o[2][r] *= al; o[3][r] *= al;
    }

    // re-layout P (C/D fragment) -> A fragment through LDS, as bf16 [16][32]
    __syncthreads();
#pragma unroll
    for (int r = 0; r < 8; ++r) {
      const int row = r + hi * 8;
      Plds[row * 32 + col]      = (__bf16)s0[r];
      Plds[row * 32 + 16 + col] = (__bf16)s1[r];
    }
    __syncthreads();
    v16bf pa = load_afrag(&Plds[col * 32], 0, hi);

    // O(16x64) += P(16x32) x V(32x64); V stored transposed -> rows of Vt contiguous
#pragma unroll
    for (int dt = 0; dt < 4; ++dt) {
      const __bf16* vrow = Vp + (size_t)(dt * 16 + col) * SEQ;
      o[dt] = wmma_bf16(pa, load_bfrag(vrow, kb, hi), o[dt]);
    }
  }

  // normalize + write attention output (bf16, [pair][b*L+l][h*64+d])
  __bf16* obase = Ob + ((size_t)pair * MTOT + (size_t)bb * SEQ) * EMB;
#pragma unroll
  for (int r = 0; r < 8; ++r) {
    const float inv = 1.0f / lrun[r];
    const int lq = qt * 16 + r + hi * 8;
    __bf16* orow = obase + (size_t)lq * EMB + h * HDIM;
#pragma unroll
    for (int dt = 0; dt < 4; ++dt)
      orow[dt * 16 + col] = (__bf16)(o[dt][r] * inv);
  }
}

// ---------------- kernel 3: output projection (32x64 register tile) ----------------
__global__ __launch_bounds__(32) void out_proj(const __bf16* __restrict__ Ob,
                                               const __bf16* __restrict__ owbf,
                                               const float* __restrict__ out_b,
                                               float* __restrict__ out) {
  const int lane = threadIdx.x;
  const int col  = lane & 15;
  const int hi   = lane >> 4;

  int t = blockIdx.x;                      // pair*128*8 + mt*8 + nt
  const int nt   = t % (EMB / 64); t /= (EMB / 64);
  const int mt   = t % (MTOT / 32);
  const int pair = t / (MTOT / 32);
  const int m0 = mt * 32, n0 = nt * 64;

  const __bf16* A0 = Ob + (size_t)pair * MTOT * EMB + (size_t)(m0 + col) * EMB;
  const __bf16* A1 = A0 + (size_t)16 * EMB;
  const __bf16* Bw = owbf + (size_t)(n0 + col) * EMB;

  v8f acc[2][4] = {};
  for (int k0 = 0; k0 < EMB; k0 += 32) {
    v16bf a0 = load_afrag(A0, k0, hi);
    v16bf a1 = load_afrag(A1, k0, hi);
#pragma unroll
    for (int sub = 0; sub < 4; ++sub) {
      v16bf b = load_bfrag(Bw + (size_t)sub * 16 * EMB, k0, hi);
      acc[0][sub] = wmma_bf16(a0, b, acc[0][sub]);
      acc[1][sub] = wmma_bf16(a1, b, acc[1][sub]);
    }
  }

  float bias[4];
#pragma unroll
  for (int sub = 0; sub < 4; ++sub) bias[sub] = out_b[n0 + sub * 16 + col];

  float* obase = out + (size_t)pair * MTOT * EMB;
#pragma unroll
  for (int mtile = 0; mtile < 2; ++mtile) {
#pragma unroll
    for (int r = 0; r < 8; ++r) {
      float* orow = obase + (size_t)(m0 + mtile * 16 + r + hi * 8) * EMB;
#pragma unroll
      for (int sub = 0; sub < 4; ++sub)
        orow[n0 + sub * 16 + col] = acc[mtile][sub][r] + bias[sub];
    }
  }
}

// ---------------- host launcher ----------------
extern "C" void kernel_launch(void* const* d_in, const int* in_sizes, int n_in,
                              void* d_out, int out_size, void* d_ws, size_t ws_size,
                              hipStream_t stream) {
  const float* x1    = (const float*)d_in[0];
  const float* x2    = (const float*)d_in[1];
  const float* qkv_w = (const float*)d_in[2];
  const float* qkv_b = (const float*)d_in[3];
  const float* out_w = (const float*)d_in[4];
  const float* out_b = (const float*)d_in[5];
  float* out = (float*)d_out;

  const size_t NX = (size_t)MTOT * EMB;    // 2,097,152 elements
  const size_t NW = (size_t)F3 * EMB;      //   786,432
  const size_t NO = (size_t)EMB * EMB;     //   262,144

  __bf16* xbf  = (__bf16*)d_ws;            // [2][4096][512]
  __bf16* wbf  = xbf + 2 * NX;             // [1536][512]
  __bf16* owbf = wbf + NW;                 // [512][512]
  __bf16* Qb   = owbf + NO;                // [2][2][8][2048][64] (scaled)
  __bf16* Kb   = Qb + 2 * NX;              // [2][2][8][2048][64]
  __bf16* Vt   = Kb + 2 * NX;              // [2][2][8][64][2048]
  __bf16* Ob   = Vt + 2 * NX;              // [2][4096][512]
  // total workspace: ~44 MB of bf16

  {
    const size_t total = 2 * NX + NW + NO;
    const int blocks = (int)((total + 255) / 256);
    pack_bf16<<<blocks, 256, 0, stream>>>(x1, x2, qkv_w, out_w, xbf, wbf, owbf);
  }
  {
    const int blocks = 2 * (MTOT / 32) * (F3 / 64);    // 6144
    qkv_gemm<<<blocks, 32, 0, stream>>>(xbf, wbf, qkv_b, Qb, Kb, Vt);
  }
  {
    const int blocks = 2 * BATCH * HEADS * (SEQ / 16); // 4096
    cross_attn<<<blocks, 32, 0, stream>>>(Qb, Kb, Vt, Ob);
  }
  {
    const int blocks = 2 * (MTOT / 32) * (EMB / 64);   // 2048
    out_proj<<<blocks, 32, 0, stream>>>(Ob, owbf, out_b, out);
  }
}
